// LanguageModelSRNN_76055280878226
// MI455X (gfx1250) — matrix operand order repaired and checked by reading, not compiled, not run on hardware
//
#include <hip/hip_runtime.h>
#include <hip/hip_bf16.h>

// ---------------------------------------------------------------------------
// LanguageModelSRNN on MI455X (gfx1250, wave32, WMMA)
//   h_t = tanh(h_{t-1} @ U^T + W[:, x_t]);  out = softmax(hs @ lin_w^T + b)
//
// Phase 1: pack U and lin_w into bf16 WMMA B-fragment layout (+ zero barriers)
// Phase 2: persistent recurrence. 4 batch-groups x 16 WGs. Each WG keeps its
//          128KB U slice resident in LDS (WGP has 320KB). Per step: 32x
//          v_wmma_f32_16x16x32_bf16 per wave (two accumulator chains), tanh
//          epilogue, h exchanged via 32KB L2 ping-pong + device-scope group
//          barrier. hs stored as bf16 directly in A-fragment layout.
// Phase 3: projection (8 WMMA acc tiles per wave over K=1024, Wpk staged in
//          LDS) fused with bias + stable softmax via wave32 shfl_xor.
// Data movement uses GLOBAL_LOAD_ASYNC_TO_LDS_B128 when the toolchain
// exposes it (ASYNCcnt path), otherwise falls back to load+ds_store.
// ---------------------------------------------------------------------------

typedef __attribute__((ext_vector_type(16))) __bf16 v16bf;
typedef __attribute__((ext_vector_type(8)))  float  v8f;
typedef int v4i __attribute__((vector_size(16)));

#define HIDDEN 1024
#define VOCAB  128
#define BATCH  64
#define SEQ    512
#define NKB    32   // K blocks of 32 along hidden

// workspace layout (bytes)
#define OFF_CNT  0                         // 4 uint barrier counters
#define OFF_UPK  256                       // 64*32*32*16 bf16 = 2 MB
#define OFF_WPK  (OFF_UPK + 2097152)       // 8*32*32*16 bf16 = 256 KB
#define OFF_HBUF (OFF_WPK + 262144)        // 4 groups * 2 * 16384 bf16 = 256 KB
#define OFF_HS   (OFF_HBUF + 262144)       // 2048*32*32*16 bf16 = 64 MB

#define AS3 __attribute__((address_space(3)))

#if __has_builtin(__builtin_amdgcn_global_load_async_to_lds_b128)
#define HAVE_ASYNC_LDS 1
#else
#define HAVE_ASYNC_LDS 0
#endif

#if HAVE_ASYNC_LDS && __has_builtin(__builtin_amdgcn_s_wait_asynccnt)
#define ASYNC_WAIT() __builtin_amdgcn_s_wait_asynccnt(0)
#elif HAVE_ASYNC_LDS
#define ASYNC_WAIT() asm volatile("s_wait_asynccnt 0x0" ::: "memory")
#else
#define ASYNC_WAIT() ((void)0)
#endif

// copy 16 bytes global -> LDS (async DMA when available)
__device__ __forceinline__ void g2l_b128(void* lds_dst, const void* gsrc) {
#if HAVE_ASYNC_LDS
  __builtin_amdgcn_global_load_async_to_lds_b128(
      (v4i*)gsrc,
      (AS3 v4i*)(unsigned)(unsigned long long)lds_dst,
      0, 0);
#else
  *(float4*)lds_dst = *(const float4*)gsrc;
#endif
}

__device__ __forceinline__ float fast_tanh(float x) {
  float ax = __builtin_fabsf(x);
  float e  = __expf(2.0f * ax);                    // v_exp_f32
  float r  = 1.0f - 2.0f * __builtin_amdgcn_rcpf(e + 1.0f);
  return x < 0.0f ? -r : r;
}

// --- Phase 1: pack weights to bf16 WMMA B-fragment layout, reset barriers ---
// B frag (32x16, bf16): lane L, half j: K = (L&16)+j, N = L&15
__global__ void k_pack(const float* __restrict__ U, const float* __restrict__ lin_w,
                       __hip_bfloat16* __restrict__ Upk, __hip_bfloat16* __restrict__ Wpk,
                       unsigned* __restrict__ cnt) {
  int t = blockIdx.x * 256 + threadIdx.x;
  if (t < 4) cnt[t] = 0u;
  if (t < 1048576) {                       // U: B[k][n] = U[n][k]
    int j = t & 15, lane = (t >> 4) & 31, kb = (t >> 9) & 31, nb = t >> 14;
    int n = nb * 16 + (lane & 15);
    int k = kb * 32 + (lane & 16) + j;
    Upk[t] = __float2bfloat16(U[n * HIDDEN + k]);
  } else if (t < 1048576 + 131072) {       // lin_w: B[k][v] = lin_w[v][k]
    int f = t - 1048576;
    int j = f & 15, lane = (f >> 4) & 31, kb = (f >> 9) & 31, vb = f >> 14;
    int v = vb * 16 + (lane & 15);
    int k = kb * 32 + (lane & 16) + j;
    Wpk[f] = __float2bfloat16(lin_w[v * HIDDEN + k]);
  }
}

// --- Phase 2: persistent recurrence -----------------------------------------
// grid = 64 blocks (4 batch groups x 16 WGs), 128 threads (4 waves)
// dynamic LDS: 128KB U slice + 32KB h A-fragments = 160KB
__global__ void k_rnn(const int* __restrict__ X, const float* __restrict__ W,
                      const __hip_bfloat16* __restrict__ Upk,
                      __hip_bfloat16* __restrict__ hbuf,
                      __hip_bfloat16* __restrict__ hs,
                      unsigned* __restrict__ cnt) {
  extern __shared__ char smem[];
  __hip_bfloat16* sU = (__hip_bfloat16*)smem;              // 65536 bf16
  __hip_bfloat16* sH = (__hip_bfloat16*)(smem + 131072);   // 16384 bf16

  const int bt  = blockIdx.x >> 4;      // batch group 0..3 (16 rows each)
  const int g   = blockIdx.x & 15;      // workgroup in group: 64 N-columns
  const int wv  = threadIdx.x >> 5;     // wave 0..3 -> one 16-wide N tile
  const int L   = threadIdx.x & 31;
  const int tid = threadIdx.x;

  // resident U slice: contiguous 128KB region for n-tiles [g*4, g*4+4)
  {
    const char* src = (const char*)(Upk + (size_t)g * 65536);
    for (int i = tid; i < 8192; i += 128)
      g2l_b128(smem + i * 16, src + (size_t)i * 16);
  }
  // h_0 = 0 in A-fragment layout
  {
    float4 z = {0.f, 0.f, 0.f, 0.f};
    float4* dz = (float4*)sH;
    for (int i = tid; i < 2048; i += 128) dz[i] = z;
  }
  ASYNC_WAIT();
  __syncthreads();

  const v16bf* sUv = (const v16bf*)sU;  // frag (ntl,kb): [(ntl*32+kb)*32 + L]
  const v16bf* sHv = (const v16bf*)sH;  // frag (kb):     [kb*32 + L]
  const int nt = g * 4 + wv;            // global N tile
  const int n  = nt * 16 + (L & 15);    // this lane's output column
  const int hi = (L >> 4);              // 0: M=r, 1: M=r+8
  unsigned* mycnt = cnt + bt;
  __hip_bfloat16* hb = hbuf + (size_t)bt * 2 * 16384;

  for (int s = 0; s < SEQ; ++s) {
    // two independent accumulation chains (even/odd K blocks)
    v8f acc0 = {}, acc1 = {};
    #pragma unroll 4
    for (int kb = 0; kb < NKB; kb += 2) {
      v16bf a0 = sHv[kb * 32 + L];
      v16bf b0 = sUv[(wv * 32 + kb) * 32 + L];
      acc0 = __builtin_amdgcn_wmma_f32_16x16x32_bf16(
          false, a0, false, b0, (short)0, acc0, false, false);
      v16bf a1 = sHv[(kb + 1) * 32 + L];
      v16bf b1 = sUv[(wv * 32 + kb + 1) * 32 + L];
      acc1 = __builtin_amdgcn_wmma_f32_16x16x32_bf16(
          false, a1, false, b1, (short)0, acc1, false, false);
    }
    // epilogue: + W[n, x_t] gather, tanh, scatter to h ping-pong + hs frags
    const int s16 = s & 15, st = s >> 4;
    const int rem = n & 31, kb2 = n >> 5;
    const int i2  = rem & 15;
    const int lsel = (rem & 16) ? 16 : 0;
    #pragma unroll
    for (int r = 0; r < 8; ++r) {
      int m = r + (hi ? 8 : 0);
      int b = bt * 16 + m;
      int tok = X[b * SEQ + s];
      float h = fast_tanh(acc0[r] + acc1[r] + W[n * VOCAB + tok]);
      __hip_bfloat16 h16 = __float2bfloat16(h);
      hb[(s & 1) * 16384 + (kb2 * 32 + (lsel + m)) * 16 + i2] = h16;
      size_t rt = (size_t)b * 32 + st;   // row tile of (b*SEQ+s)
      hs[((rt * 32 + kb2) * 32 + (lsel + s16)) * 16 + i2] = h16;
    }
    // device-scope group barrier (16 WGs per batch group)
    __threadfence();
    __syncthreads();
    if (tid == 0) {
      __hip_atomic_fetch_add(mycnt, 1u, __ATOMIC_RELEASE, __HIP_MEMORY_SCOPE_AGENT);
      unsigned target = 16u * (unsigned)(s + 1);
      while (__hip_atomic_load(mycnt, __ATOMIC_RELAXED, __HIP_MEMORY_SCOPE_AGENT) < target)
        __builtin_amdgcn_s_sleep(1);
      __threadfence();
    }
    __syncthreads();
    // pull full h_t (32KB) from L2 ping-pong into LDS A-fragments (async DMA)
    {
      const char* src = (const char*)(hb + (s & 1) * 16384);
      for (int i = tid; i < 2048; i += 128)
        g2l_b128((char*)sH + i * 16, src + (size_t)i * 16);
    }
    ASYNC_WAIT();
    __syncthreads();
  }
}

// --- Phase 3: projection + bias + softmax ------------------------------------
// grid = 256 blocks x 256 threads; one wave per 16-row tile (2048 tiles)
// dynamic LDS: full Wpk staged (256KB)
__global__ void k_proj(const __hip_bfloat16* __restrict__ hs,
                       const __hip_bfloat16* __restrict__ Wpk,
                       const float* __restrict__ lin_b,
                       float* __restrict__ out) {
  extern __shared__ char smem[];
  const int tid = threadIdx.x;
  // stage all lin_w fragments into LDS once per workgroup
  for (int i = tid; i < 16384; i += 256)
    g2l_b128(smem + i * 16, (const char*)Wpk + (size_t)i * 16);
  ASYNC_WAIT();
  __syncthreads();

  const int wvv = threadIdx.x >> 5, L = threadIdx.x & 31;
  const int rt = blockIdx.x * 8 + wvv;
  const v16bf* A  = (const v16bf*)hs + (size_t)rt * 32 * 32;
  const v16bf* Bp = (const v16bf*)smem;

  v8f acc[8] = {{}, {}, {}, {}, {}, {}, {}, {}};
  for (int kb = 0; kb < NKB; ++kb) {
    v16bf a = A[kb * 32 + L];
    #pragma unroll
    for (int vb = 0; vb < 8; ++vb) {
      v16bf b = Bp[(vb * 32 + kb) * 32 + L];
      acc[vb] = __builtin_amdgcn_wmma_f32_16x16x32_bf16(
          false, a, false, b, (short)0, acc[vb], false, false);
    }
  }
  float bias[8];
  #pragma unroll
  for (int vb = 0; vb < 8; ++vb) bias[vb] = lin_b[vb * 16 + (L & 15)];

  const int half = L >> 4;
  #pragma unroll
  for (int r = 0; r < 8; ++r) {
    float mx = -3.0e38f;
    #pragma unroll
    for (int vb = 0; vb < 8; ++vb) {
      float v = acc[vb][r] + bias[vb];
      acc[vb][r] = v;
      mx = fmaxf(mx, v);
    }
    #pragma unroll
    for (int d = 1; d < 16; d <<= 1) mx = fmaxf(mx, __shfl_xor(mx, d, 32));
    float sum = 0.f;
    #pragma unroll
    for (int vb = 0; vb < 8; ++vb) {
      float e = __expf(acc[vb][r] - mx);
      acc[vb][r] = e;
      sum += e;
    }
    #pragma unroll
    for (int d = 1; d < 16; d <<= 1) sum += __shfl_xor(sum, d, 32);
    float inv = __builtin_amdgcn_rcpf(sum);
    size_t row = (size_t)rt * 16 + (r + half * 8);
    #pragma unroll
    for (int vb = 0; vb < 8; ++vb)
      out[row * VOCAB + vb * 16 + (L & 15)] = acc[vb][r] * inv;
  }
}

extern "C" void kernel_launch(void* const* d_in, const int* in_sizes, int n_in,
                              void* d_out, int out_size, void* d_ws, size_t ws_size,
                              hipStream_t stream) {
  const int*   X     = (const int*)d_in[0];
  const float* U     = (const float*)d_in[1];
  const float* W     = (const float*)d_in[2];
  const float* lin_w = (const float*)d_in[3];
  const float* lin_b = (const float*)d_in[4];
  char* ws = (char*)d_ws;
  unsigned*       cnt  = (unsigned*)(ws + OFF_CNT);
  __hip_bfloat16* Upk  = (__hip_bfloat16*)(ws + OFF_UPK);
  __hip_bfloat16* Wpk  = (__hip_bfloat16*)(ws + OFF_WPK);
  __hip_bfloat16* hbuf = (__hip_bfloat16*)(ws + OFF_HBUF);
  __hip_bfloat16* hs   = (__hip_bfloat16*)(ws + OFF_HS);
  float* out = (float*)d_out;

  k_pack<<<4608, 256, 0, stream>>>(U, lin_w, Upk, Wpk, cnt);
  k_rnn<<<64, 128, 160 * 1024, stream>>>(X, W, Upk, hbuf, hs, cnt);
  k_proj<<<256, 256, 262144, stream>>>(hs, Wpk, lin_b, out);
}
